// SqueezeBodyEdge_3676492005661
// MI455X (gfx1250) — compile-verified, matrix operand order checked by reading, not run on hardware
//
#include <hip/hip_runtime.h>
#include <hip/hip_bf16.h>
#include <math.h>

// ---------------------------------------------------------------------------
// MI455X (gfx1250, wave32) implementation.
// ResNet convs -> implicit GEMM on v_wmma_f32_16x16x32_f16 (f32 accumulate).
// Block tile 128x64, K-step 32, 8 waves (4x2), each wave = 2x2 WMMA tiles.
// Register double-buffered staging, b128 LDS fragment loads, b128 stores.
// ---------------------------------------------------------------------------

typedef _Float16 f16;
typedef __attribute__((ext_vector_type(16))) _Float16 v16h;
typedef __attribute__((ext_vector_type(8)))  _Float16 v8h;
typedef __attribute__((ext_vector_type(8)))  float    v8f;

#define BM 128
#define BN 64
#define BK 32
#define LSTRIDE 40   // padded LDS row stride in halves (80B: 16B aligned, bank-skewed)

// ---------------------------------------------------------------------------
// Weight transpose + f16 convert: w[co][k] (OIHW flattened) -> wT[k][co]
// ---------------------------------------------------------------------------
__global__ void transpose_w_kernel(const float* __restrict__ w, f16* __restrict__ wT,
                                   int Cout, int K) {
  int i = blockIdx.x * 256 + threadIdx.x;
  if (i >= Cout * K) return;
  int co = i / K;
  int k  = i % K;
  wT[(size_t)k * Cout + co] = (f16)w[i];
}

// ---------------------------------------------------------------------------
// Implicit-GEMM conv at fixed 128x128 spatial, KSZ in {3,1}.
//   out = gamma*(conv(in)+bias) + beta  [+ resid]  [relu]
// M = B*128*128 = 32768 rows, N = Cout (multiple of 64).
// ---------------------------------------------------------------------------
template <int KSZ, int PAD>
__global__ __launch_bounds__(256) void conv_wmma_kernel(
    const float* __restrict__ in, const f16* __restrict__ wT,
    const float* __restrict__ bias, const float* __restrict__ gamma,
    const float* __restrict__ beta, const float* __restrict__ resid,
    float* __restrict__ out, int Cin, int Cout, int relu)
{
  constexpr int H = 128, W = 128, HW = H * W;          // HW = 16384
  __shared__ f16 As [BM * LSTRIDE];                    // 10 KB, [m][k] padded
  __shared__ f16 Bst[BN * LSTRIDE];                    // 5 KB,  [n][k] padded
  const int Ktot = Cin * KSZ * KSZ;
  const int m0   = blockIdx.x * BM;
  const int n0   = blockIdx.y * BN;
  const int tid  = threadIdx.x;
  const int wave = tid >> 5, lane = tid & 31;
  const int wm   = wave & 3,  wn   = wave >> 2;        // 4x2 wave grid
  const int half = lane >> 4, r    = lane & 15;

  // fixed per-thread staging slots
  const int ka      = tid & 31;    // A: fixed k column; rows ma_base + 8*t
  const int ma_base = tid >> 5;
  const int nb      = tid & 63;    // B: fixed n column; rows kb_base + 4*t
  const int kb_base = tid >> 6;

  v8f acc[2][2] = {};
  f16 va[16];
  f16 vb[8];

  auto stageA = [&](int k0) {
    int kg = k0 + ka;
    if (kg < Ktot) {
      int ci, ky, kx;
      if (KSZ == 1) { ci = kg; ky = 0; kx = 0; }
      else { ci = kg / 9; int rem = kg % 9; ky = rem / 3; kx = rem % 3; }
      const float* cbase = in + (size_t)ci * HW;
#pragma unroll
      for (int t = 0; t < 16; ++t) {
        int mg = m0 + ma_base + 8 * t;
        int b  = mg >> 14;              // HW == 16384
        int hw = mg & (HW - 1);
        int y  = hw >> 7, xx = hw & (W - 1);
        int iy = y + ky - PAD, ix = xx + kx - PAD;
        f16 v = (f16)0.f;
        if ((unsigned)iy < (unsigned)H && (unsigned)ix < (unsigned)W)
          v = (f16)cbase[(size_t)b * Cin * HW + iy * W + ix];
        va[t] = v;
      }
    } else {
#pragma unroll
      for (int t = 0; t < 16; ++t) va[t] = (f16)0.f;
    }
  };
  auto stageB = [&](int k0) {
#pragma unroll
    for (int t = 0; t < 8; ++t) {
      int kg = k0 + kb_base + 4 * t;
      vb[t] = (kg < Ktot) ? wT[(size_t)kg * Cout + n0 + nb] : (f16)0.f;
    }
  };
  auto commit = [&]() {
#pragma unroll
    for (int t = 0; t < 16; ++t) As[(ma_base + 8 * t) * LSTRIDE + ka] = va[t];
#pragma unroll
    for (int t = 0; t < 8; ++t)  Bst[nb * LSTRIDE + kb_base + 4 * t] = vb[t];
  };

  stageA(0);
  stageB(0);
  for (int k0 = 0; k0 < Ktot; k0 += BK) {
    commit();
    __syncthreads();

    int kn = k0 + BK;
    if (kn < Ktot) {
      // pull next weight tile toward L2 two steps ahead (global_prefetch_b8)
      if (kn + BK < Ktot)
        __builtin_prefetch(&wT[(size_t)(kn + BK) * Cout + n0 + nb], 0, 1);
      stageA(kn);    // overlap global gather with WMMA below
      stageB(kn);
    }

    // fragments via two ds_load_b128 each (16-bit A 16x32 lane layout:
    // lane half selects K 0..7 / 8..15 in [0,16) and 16..23 / 24..31 in [16,32))
    v16h afrag[2], bfrag[2];
#pragma unroll
    for (int mi = 0; mi < 2; ++mi) {
      const f16* rp = &As[(wm * 32 + mi * 16 + r) * LSTRIDE + half * 8];
      v8h lo = *(const v8h*)rp;
      v8h hi = *(const v8h*)(rp + 16);
      afrag[mi] = __builtin_shufflevector(lo, hi, 0, 1, 2, 3, 4, 5, 6, 7,
                                          8, 9, 10, 11, 12, 13, 14, 15);
    }
#pragma unroll
    for (int ni = 0; ni < 2; ++ni) {
      const f16* cp = &Bst[(wn * 32 + ni * 16 + r) * LSTRIDE + half * 8];
      v8h lo = *(const v8h*)cp;
      v8h hi = *(const v8h*)(cp + 16);
      bfrag[ni] = __builtin_shufflevector(lo, hi, 0, 1, 2, 3, 4, 5, 6, 7,
                                          8, 9, 10, 11, 12, 13, 14, 15);
    }
#pragma unroll
    for (int mi = 0; mi < 2; ++mi)
#pragma unroll
      for (int ni = 0; ni < 2; ++ni)
        acc[mi][ni] = __builtin_amdgcn_wmma_f32_16x16x32_f16(
            false, afrag[mi], false, bfrag[ni], (short)0, acc[mi][ni], false, false);
    __syncthreads();
  }

  // ---- epilogue: (acc+bias)*gamma+beta [+resid] [relu], 2x b128 stores ----
#pragma unroll
  for (int mi = 0; mi < 2; ++mi) {
    int mg0 = m0 + wm * 32 + mi * 16 + half * 8;   // 8 consecutive spatial rows
    int b   = mg0 >> 14;
    int hw0 = mg0 & (HW - 1);
#pragma unroll
    for (int ni = 0; ni < 2; ++ni) {
      int co = n0 + wn * 32 + ni * 16 + r;
      float bi = bias[co], gm = gamma[co], bt = beta[co];
      size_t ob = ((size_t)(b * Cout + co)) * HW + hw0;
      float tmp[8];
#pragma unroll
      for (int j = 0; j < 8; ++j) tmp[j] = (acc[mi][ni][j] + bi) * gm + bt;
      if (resid) {
        float4 a0 = *(const float4*)(resid + ob);
        float4 a1 = *(const float4*)(resid + ob + 4);
        tmp[0] += a0.x; tmp[1] += a0.y; tmp[2] += a0.z; tmp[3] += a0.w;
        tmp[4] += a1.x; tmp[5] += a1.y; tmp[6] += a1.z; tmp[7] += a1.w;
      }
      if (relu) {
#pragma unroll
        for (int j = 0; j < 8; ++j) tmp[j] = fmaxf(tmp[j], 0.f);
      }
      float4 s0 = {tmp[0], tmp[1], tmp[2], tmp[3]};
      float4 s1 = {tmp[4], tmp[5], tmp[6], tmp[7]};
      *(float4*)(out + ob)     = s0;
      *(float4*)(out + ob + 4) = s1;
    }
  }
}

// ---------------------------------------------------------------------------
// normalize_images: per (b,c) plane, one block each.
// ---------------------------------------------------------------------------
__global__ void normalize_images_kernel(const float* __restrict__ in,
                                        float* __restrict__ out, int HW) {
  const float* p = in + (size_t)blockIdx.x * HW;
  float* o = out + (size_t)blockIdx.x * HW;
  __shared__ float r0[256], r1[256];
  int tid = threadIdx.x;
  float s = 0.f, s2 = 0.f;
  for (int i = tid; i < HW; i += 256) { float v = p[i]; s += v; s2 += v * v; }
  r0[tid] = s; r1[tid] = s2; __syncthreads();
  for (int off = 128; off; off >>= 1) {
    if (tid < off) { r0[tid] += r0[tid + off]; r1[tid] += r1[tid + off]; }
    __syncthreads();
  }
  float mean = r0[0] / HW;
  float var  = r1[0] / HW - mean * mean;
  float sd   = sqrtf(fmaxf(var, 0.f)) + 1e-8f;
  __syncthreads();
  float mn = 1e30f, mx = -1e30f;
  for (int i = tid; i < HW; i += 256) {
    float nv = (p[i] - mean) / sd;
    mn = fminf(mn, nv); mx = fmaxf(mx, nv);
  }
  r0[tid] = mn; r1[tid] = mx; __syncthreads();
  for (int off = 128; off; off >>= 1) {
    if (tid < off) { r0[tid] = fminf(r0[tid], r0[tid + off]); r1[tid] = fmaxf(r1[tid], r1[tid + off]); }
    __syncthreads();
  }
  float lo = r0[0], rng = r1[0] - r0[0];
  for (int i = tid; i < HW; i += 256)
    o[i] = ((p[i] - mean) / sd - lo) / rng;
}

// ---------------------------------------------------------------------------
// Depthwise 3x3 stride-2 pad-0 conv + BN + ReLU (groups == C).
// ---------------------------------------------------------------------------
__global__ void dwconv_s2_kernel(const float* __restrict__ in, const float* __restrict__ w,
                                 const float* __restrict__ g, const float* __restrict__ b,
                                 float* __restrict__ out, int C, int Hin, int Win,
                                 int Hout, int Wout, int total) {
  int idx = blockIdx.x * 256 + threadIdx.x;
  if (idx >= total) return;
  int x = idx % Wout, y = (idx / Wout) % Hout;
  int c = (idx / (Wout * Hout)) % C, bb = idx / (Wout * Hout * C);
  const float* src = in + ((size_t)(bb * C + c)) * Hin * Win;
  const float* wp  = w + c * 9;
  float acc = 0.f;
  for (int ky = 0; ky < 3; ++ky)
    for (int kx = 0; kx < 3; ++kx)
      acc += src[(2 * y + ky) * Win + (2 * x + kx)] * wp[ky * 3 + kx];
  acc = acc * g[c] + b[c];
  out[idx] = fmaxf(acc, 0.f);
}

// ---------------------------------------------------------------------------
// Align-corners bilinear resize.
// ---------------------------------------------------------------------------
__global__ void resize_ac_kernel(const float* __restrict__ in, float* __restrict__ out,
                                 int C, int Hin, int Win, int Hout, int Wout, int total) {
  int idx = blockIdx.x * 256 + threadIdx.x;
  if (idx >= total) return;
  int x = idx % Wout, y = (idx / Wout) % Hout;
  int plane = idx / (Wout * Hout);   // b*C + c
  float ys = (float)y * (float)(Hin - 1) / (float)(Hout - 1);
  float xs = (float)x * (float)(Win - 1) / (float)(Wout - 1);
  int y0 = (int)floorf(ys), x0 = (int)floorf(xs);
  int y1 = min(y0 + 1, Hin - 1), x1 = min(x0 + 1, Win - 1);
  float wy = ys - y0, wx = xs - x0;
  const float* p = in + (size_t)plane * Hin * Win;
  float top = (1.f - wx) * p[y0 * Win + x0] + wx * p[y0 * Win + x1];
  float bot = (1.f - wx) * p[y1 * Win + x0] + wx * p[y1 * Win + x1];
  out[idx] = (1.f - wy) * top + wy * bot;
}

// ---------------------------------------------------------------------------
// Direct 3x3 pad-1 conv over concatenated inputs [inA(Ca) ; inB(Cb)], no bias.
// ---------------------------------------------------------------------------
__global__ void conv3x3_cat_kernel(const float* __restrict__ inA, int Ca,
                                   const float* __restrict__ inB, int Cb,
                                   const float* __restrict__ w, float* __restrict__ out,
                                   int Cout, int H, int W, int total) {
  int idx = blockIdx.x * 256 + threadIdx.x;
  if (idx >= total) return;
  int x = idx % W, y = (idx / W) % H;
  int co = (idx / (W * H)) % Cout, b = idx / (W * H * Cout);
  int Cin = Ca + Cb;
  float acc = 0.f;
  for (int ci = 0; ci < Cin; ++ci) {
    const float* src = (ci < Ca)
        ? inA + ((size_t)(b * Ca + ci)) * H * W
        : inB + ((size_t)(b * Cb + (ci - Ca))) * H * W;
    const float* wp = w + ((size_t)(co * Cin + ci)) * 9;
    for (int ky = 0; ky < 3; ++ky) {
      int iy = y + ky - 1;
      if ((unsigned)iy >= (unsigned)H) continue;
      for (int kx = 0; kx < 3; ++kx) {
        int ix = x + kx - 1;
        if ((unsigned)ix >= (unsigned)W) continue;
        acc += src[iy * W + ix] * wp[ky * 3 + kx];
      }
    }
  }
  out[idx] = acc;
}

// ---------------------------------------------------------------------------
// Flow-grid warp (zero-pad bilinear grid_sample) + edge = x - warp.
// ---------------------------------------------------------------------------
__global__ void warp_edge_kernel(const float* __restrict__ xn, const float* __restrict__ flow,
                                 float* __restrict__ warp, float* __restrict__ edge,
                                 int C, int H, int W, int total) {
  int idx = blockIdx.x * 256 + threadIdx.x;
  if (idx >= total) return;               // total = B*H*W
  int x = idx % W, y = (idx / W) % H, b = idx / (W * H);
  float wg = -1.f + 2.f * x / (float)(W - 1);
  float hg = -1.f + 2.f * y / (float)(H - 1);
  float gxn = wg + flow[((size_t)(b * 2 + 0)) * H * W + y * W + x] / (float)W;
  float gyn = hg + flow[((size_t)(b * 2 + 1)) * H * W + y * W + x] / (float)H;
  float gx = (gxn + 1.f) * W * 0.5f - 0.5f;
  float gy = (gyn + 1.f) * H * 0.5f - 0.5f;
  float x0f = floorf(gx), y0f = floorf(gy);
  float fx = gx - x0f, fy = gy - y0f;
  int xi0 = (int)x0f, yi0 = (int)y0f;
  for (int c = 0; c < C; ++c) {
    const float* p = xn + ((size_t)(b * C + c)) * H * W;
    float v[4];
    int dx[4] = {0, 1, 0, 1}, dy[4] = {0, 0, 1, 1};
    for (int q = 0; q < 4; ++q) {
      int xi = xi0 + dx[q], yi = yi0 + dy[q];
      bool ok = (xi >= 0) && (xi <= W - 1) && (yi >= 0) && (yi <= H - 1);
      int xc = min(max(xi, 0), W - 1), yc = min(max(yi, 0), H - 1);
      v[q] = ok ? p[yc * W + xc] : 0.f;
    }
    float s = (1.f - fx) * (1.f - fy) * v[0] + fx * (1.f - fy) * v[1]
            + (1.f - fx) * fy * v[2] + fx * fy * v[3];
    size_t o = ((size_t)(b * C + c)) * H * W + y * W + x;
    warp[o] = s;
    edge[o] = p[y * W + x] - s;
  }
}

// ---------------------------------------------------------------------------
// Sobel magnitude threshold (per plane, zero pad).
// ---------------------------------------------------------------------------
__global__ void sobel_kernel(const float* __restrict__ in, float* __restrict__ out,
                             int H, int W, int total) {
  int idx = blockIdx.x * 256 + threadIdx.x;
  if (idx >= total) return;
  int x = idx % W, y = (idx / W) % H;
  int plane = idx / (W * H);
  const float* p = in + (size_t)plane * H * W;
  const float KX[9] = {1.f, 0.f, -1.f, 2.f, 0.f, -2.f, 1.f, 0.f, -1.f};
  float gx = 0.f, gy = 0.f;
  for (int ky = 0; ky < 3; ++ky) {
    int iy = y + ky - 1;
    if ((unsigned)iy >= (unsigned)H) continue;
    for (int kx = 0; kx < 3; ++kx) {
      int ix = x + kx - 1;
      if ((unsigned)ix >= (unsigned)W) continue;
      float v = p[iy * W + ix];
      gx += KX[ky * 3 + kx] * v;
      gy += KX[kx * 3 + ky] * v;   // transpose of KX
    }
  }
  float mag = sqrtf(gx * gx + gy * gy + 1e-12f);
  out[idx] = (mag > 1.f) ? 1.f : 0.f;
}

// ---------------------------------------------------------------------------
// MSE((em[b,0]) - es[b,3]) over a 128x128 plane, per batch.
// ---------------------------------------------------------------------------
__global__ void mse_kernel(const float* __restrict__ em, const float* __restrict__ es,
                           float* __restrict__ out2, int HW) {
  int b = blockIdx.x;
  __shared__ float red[256];
  int tid = threadIdx.x;
  const float* e1 = em + (size_t)b * HW;
  const float* e2 = es + ((size_t)(b * 4 + 3)) * HW;
  float s = 0.f;
  for (int i = tid; i < HW; i += 256) { float d = e1[i] - e2[i]; s += d * d; }
  red[tid] = s; __syncthreads();
  for (int off = 128; off; off >>= 1) {
    if (tid < off) red[tid] += red[tid + off];
    __syncthreads();
  }
  if (tid == 0) out2[b] = red[0] / HW;
}

// ---------------------------------------------------------------------------
// weighted = msf*wm + pan*wp, with wm/wp from tanh-normalized MSEs.
// ---------------------------------------------------------------------------
__global__ void weighted_kernel(const float* __restrict__ msf, const float* __restrict__ pan,
                                const float* __restrict__ mse_m, const float* __restrict__ mse_p,
                                float* __restrict__ out, int HW, int total) {
  int idx = blockIdx.x * 256 + threadIdx.x;
  if (idx >= total) return;
  int b = idx / HW;
  float wm = tanhf(mse_m[b]), wp = tanhf(mse_p[b]);
  float s = wm + wp + 1e-12f;
  out[idx] = msf[idx] * (wm / s) + pan[idx] * (wp / s);
}

__global__ void add_kernel(const float* __restrict__ a, const float* __restrict__ b,
                           float* __restrict__ o, int n) {
  int i = blockIdx.x * 256 + threadIdx.x;
  if (i < n) o[i] = a[i] + b[i];
}

// ---------------------------------------------------------------------------
// Global average pool: one block per (b,c) plane.
// ---------------------------------------------------------------------------
__global__ void gap_kernel(const float* __restrict__ in, float* __restrict__ pooled, int HW) {
  __shared__ float red[256];
  int tid = threadIdx.x;
  const float* p = in + (size_t)blockIdx.x * HW;
  float s = 0.f;
  for (int i = tid; i < HW; i += 256) s += p[i];
  red[tid] = s; __syncthreads();
  for (int off = 128; off; off >>= 1) {
    if (tid < off) red[tid] += red[tid + off];
    __syncthreads();
  }
  if (tid == 0) pooled[blockIdx.x] = red[0] / HW;
}

// ---------------------------------------------------------------------------
// FC (512 -> 11) + sigmoid: one block per (b, class).
// ---------------------------------------------------------------------------
__global__ void fc_sigmoid_kernel(const float* __restrict__ pooled, const float* __restrict__ fw,
                                  const float* __restrict__ fb, float* __restrict__ out,
                                  int Cin, int Ncls) {
  int b = blockIdx.x / Ncls, j = blockIdx.x % Ncls;
  __shared__ float red[128];
  int tid = threadIdx.x;
  float s = 0.f;
  for (int i = tid; i < Cin; i += 128) s += pooled[b * Cin + i] * fw[j * Cin + i];
  red[tid] = s; __syncthreads();
  for (int off = 64; off; off >>= 1) {
    if (tid < off) red[tid] += red[tid + off];
    __syncthreads();
  }
  if (tid == 0) out[b * Ncls + j] = 1.f / (1.f + expf(-(red[0] + fb[j])));
}

// ---------------------------------------------------------------------------
// Host launcher
// ---------------------------------------------------------------------------
extern "C" void kernel_launch(void* const* d_in, const int* in_sizes, int n_in,
                              void* d_out, int out_size, void* d_ws, size_t ws_size,
                              hipStream_t stream) {
  (void)in_sizes; (void)n_in; (void)out_size; (void)ws_size;
  const float* x_in    = (const float*)d_in[0];
  const float* msf     = (const float*)d_in[1];
  const float* pan     = (const float*)d_in[2];
  const float* down_w1 = (const float*)d_in[3];
  const float* down_g1 = (const float*)d_in[4];
  const float* down_b1 = (const float*)d_in[5];
  const float* down_w2 = (const float*)d_in[6];
  const float* down_g2 = (const float*)d_in[7];
  const float* down_b2 = (const float*)d_in[8];
  const float* flow_w  = (const float*)d_in[9];
  const float* c54_w   = (const float*)d_in[10];
  const float* rc1_w   = (const float*)d_in[11];
  const float* rc1_b   = (const float*)d_in[12];
  const float* rbn1_g  = (const float*)d_in[13];
  const float* rbn1_b  = (const float*)d_in[14];
  const float* fc_w    = (const float*)d_in[51];
  const float* fc_b    = (const float*)d_in[52];
  float* out = (float*)d_out;

  const int B = 2, C = 4, H = 64, W = 64, H2 = 128, W2 = 128;
  const int HW = H * W, HW2 = H2 * W2;
  const int Msp = B * H2 * W2;            // 32768 GEMM rows for ResNet
  const int PLANE4 = B * C * HW2;         // 131072

  // ---- workspace carve-out ----
  char* w8 = (char*)d_ws;
  size_t cur = 0;
  auto af = [&](size_t n) -> float* {
    float* p = (float*)(w8 + cur); cur += ((n * 4) + 255) & ~(size_t)255; return p;
  };
  auto ah = [&](size_t n) -> f16* {
    f16* p = (f16*)(w8 + cur); cur += ((n * 2) + 255) & ~(size_t)255; return p;
  };

  float* xn      = af(B * C * HW);
  float* dd1     = af(B * C * 31 * 31);
  float* dd2     = af(B * C * 15 * 15);
  float* segdown = af(B * C * HW);
  float* flow    = af(B * 2 * HW);
  float* warp    = af(B * C * HW);
  float* edge    = af(B * C * HW);
  float* nm      = af(B * HW2);
  float* npn     = af(B * HW2);
  float* ne      = af(PLANE4);
  float* em      = af(B * HW2);
  float* ep      = af(B * HW2);
  float* es      = af(PLANE4);
  float* wtd     = af(B * HW2);
  float* fine    = af(PLANE4);
  float* finalb  = af(PLANE4);
  float* pooled  = af(B * 512);
  float* bufA    = af((size_t)B * 512 * HW2);
  float* bufB    = af((size_t)B * 512 * HW2);
  float* bufC    = af((size_t)B * 512 * HW2);

  // f16 K-major weights for the WMMA convs
  f16* wt_c1 = ah(36 * 64);
  struct Blk { int ci, co; f16 *c1, *c2, *ex; int base; };
  Blk blk[3] = { {64, 128, nullptr, nullptr, nullptr, 15},
                 {128, 256, nullptr, nullptr, nullptr, 27},
                 {256, 512, nullptr, nullptr, nullptr, 39} };
  for (int i = 0; i < 3; ++i) {
    blk[i].c1 = ah((size_t)blk[i].ci * 9 * blk[i].co);
    blk[i].c2 = ah((size_t)blk[i].co * 9 * blk[i].co);
    blk[i].ex = ah((size_t)blk[i].ci * blk[i].co);
  }

  auto tr = [&](const float* w, f16* dst, int Cout, int K) {
    int n = Cout * K;
    transpose_w_kernel<<<(n + 255) / 256, 256, 0, stream>>>(w, dst, Cout, K);
  };
  tr(rc1_w, wt_c1, 64, 36);
  for (int i = 0; i < 3; ++i) {
    tr((const float*)d_in[blk[i].base + 0], blk[i].c1, blk[i].co, blk[i].ci * 9);
    tr((const float*)d_in[blk[i].base + 4], blk[i].c2, blk[i].co, blk[i].co * 9);
    tr((const float*)d_in[blk[i].base + 8], blk[i].ex, blk[i].co, blk[i].ci);
  }

  // ---- front end ----
  normalize_images_kernel<<<B * C, 256, 0, stream>>>(x_in, xn, HW);

  int t1 = B * C * 31 * 31;
  dwconv_s2_kernel<<<(t1 + 255) / 256, 256, 0, stream>>>(xn, down_w1, down_g1, down_b1,
                                                         dd1, C, H, W, 31, 31, t1);
  int t2 = B * C * 15 * 15;
  dwconv_s2_kernel<<<(t2 + 255) / 256, 256, 0, stream>>>(dd1, down_w2, down_g2, down_b2,
                                                         dd2, C, 31, 31, 15, 15, t2);
  int t3 = B * C * HW;
  resize_ac_kernel<<<(t3 + 255) / 256, 256, 0, stream>>>(dd2, segdown, C, 15, 15, H, W, t3);

  int tf = B * 2 * HW;
  conv3x3_cat_kernel<<<(tf + 255) / 256, 256, 0, stream>>>(xn, C, segdown, C, flow_w,
                                                           flow, 2, H, W, tf);
  int tw = B * HW;
  warp_edge_kernel<<<(tw + 255) / 256, 256, 0, stream>>>(xn, flow, warp, edge, C, H, W, tw);

  // upsample directly into the output tuple slots (also used downstream)
  float* sfw_up = out;                 // seg_flow_warp [2,4,128,128]
  float* se_up  = out + PLANE4;        // seg_edge      [2,4,128,128]
  float* mse_m  = out + 2 * PLANE4;    // [2]
  float* mse_p  = out + 2 * PLANE4 + 2;
  float* cls    = out + 2 * PLANE4 + 4;
  resize_ac_kernel<<<(PLANE4 + 255) / 256, 256, 0, stream>>>(warp, sfw_up, C, H, W, H2, W2, PLANE4);
  resize_ac_kernel<<<(PLANE4 + 255) / 256, 256, 0, stream>>>(edge, se_up, C, H, W, H2, W2, PLANE4);

  // canny_weight
  normalize_images_kernel<<<B, 256, 0, stream>>>(msf, nm, HW2);
  normalize_images_kernel<<<B, 256, 0, stream>>>(pan, npn, HW2);
  normalize_images_kernel<<<B * C, 256, 0, stream>>>(se_up, ne, HW2);
  int ts1 = B * HW2;
  sobel_kernel<<<(ts1 + 255) / 256, 256, 0, stream>>>(nm, em, H2, W2, ts1);
  sobel_kernel<<<(ts1 + 255) / 256, 256, 0, stream>>>(npn, ep, H2, W2, ts1);
  sobel_kernel<<<(PLANE4 + 255) / 256, 256, 0, stream>>>(ne, es, H2, W2, PLANE4);
  mse_kernel<<<B, 256, 0, stream>>>(em, es, mse_m, HW2);
  mse_kernel<<<B, 256, 0, stream>>>(ep, es, mse_p, HW2);
  weighted_kernel<<<(ts1 + 255) / 256, 256, 0, stream>>>(msf, pan, mse_m, mse_p, wtd, HW2, ts1);

  conv3x3_cat_kernel<<<(PLANE4 + 255) / 256, 256, 0, stream>>>(se_up, C, wtd, 1, c54_w,
                                                               fine, C, H2, W2, PLANE4);
  add_kernel<<<(PLANE4 + 255) / 256, 256, 0, stream>>>(sfw_up, fine, finalb, PLANE4);

  // ---- ResNet trunk via WMMA implicit GEMM ----
  auto conv3 = [&](const float* in, const f16* wT, const float* bias, const float* g,
                   const float* be, const float* resid, float* o,
                   int Cin, int Cout, int relu) {
    dim3 grid(Msp / BM, Cout / BN);
    conv_wmma_kernel<3, 1><<<grid, 256, 0, stream>>>(in, wT, bias, g, be, resid, o,
                                                     Cin, Cout, relu);
  };
  auto conv1x1 = [&](const float* in, const f16* wT, const float* bias, const float* g,
                     const float* be, const float* resid, float* o,
                     int Cin, int Cout, int relu) {
    dim3 grid(Msp / BM, Cout / BN);
    conv_wmma_kernel<1, 0><<<grid, 256, 0, stream>>>(in, wT, bias, g, be, resid, o,
                                                     Cin, Cout, relu);
  };

  // conv1: 4 -> 64, bias + bn + relu
  conv3(finalb, wt_c1, rc1_b, rbn1_g, rbn1_b, nullptr, bufA, 4, 64, 1);

  float* xb = bufA;   // block input
  float* tm = bufB;   // t1 scratch, then block output
  float* rs = bufC;   // residual branch scratch
  for (int i = 0; i < 3; ++i) {
    int bi = blk[i].base;
    const float* c1b = (const float*)d_in[bi + 1];
    const float* g1  = (const float*)d_in[bi + 2];
    const float* be1 = (const float*)d_in[bi + 3];
    const float* c2b = (const float*)d_in[bi + 5];
    const float* g2  = (const float*)d_in[bi + 6];
    const float* be2 = (const float*)d_in[bi + 7];
    const float* exb = (const float*)d_in[bi + 9];
    const float* exg = (const float*)d_in[bi + 10];
    const float* exbe= (const float*)d_in[bi + 11];
    // t1 = relu(bn(conv3x3(x)))
    conv3(xb, blk[i].c1, c1b, g1, be1, nullptr, tm, blk[i].ci, blk[i].co, 1);
    // t2 = bn(conv3x3(t1))
    conv3(tm, blk[i].c2, c2b, g2, be2, nullptr, rs, blk[i].co, blk[i].co, 0);
    // out = bn(conv1x1(x)) + t2
    conv1x1(xb, blk[i].ex, exb, exg, exbe, rs, tm, blk[i].ci, blk[i].co, 0);
    // rotate: next x = tm ; reuse old xb as scratch
    float* old = xb; xb = tm; tm = rs; rs = old;
  }

  gap_kernel<<<B * 512, 256, 0, stream>>>(xb, pooled, HW2);
  fc_sigmoid_kernel<<<B * 11, 128, 0, stream>>>(pooled, fc_w, fc_b, cls, 512, 11);
}
